// Int8OPTAttention_6141803233902
// MI455X (gfx1250) — 8-bit matrix operand order repaired and checked by reading, compiled, not run on hardware
//
#include <hip/hip_runtime.h>
#include <hip/hip_bf16.h>


typedef __attribute__((ext_vector_type(8))) int v8i;

#define E_DIM   4096
#define S_LEN   1024
#define H_NUM   32
#define DH_DIM  128
#define M_ROWS  2048
#define F32_MIN_CONST (-3.4028234663852886e38f)

__device__ __forceinline__ v8i zero8() { v8i z = {}; return z; }

__device__ __forceinline__ v8i wmma_iu8(v8i a, v8i b, v8i c) {
  return __builtin_amdgcn_wmma_i32_16x16x64_iu8(true, a, true, b, c, false, false);
}

__device__ __forceinline__ v8i load_a16x64(const signed char* base, int lda, int lane) {
  const int m  = lane & 15;
  const int kh = (lane >> 4) << 3;
  const signed char* row = base + m * lda + kh;
  int2 d0 = *(const int2*)(row);
  int2 d1 = *(const int2*)(row + 16);
  int2 d2 = *(const int2*)(row + 32);
  int2 d3 = *(const int2*)(row + 48);
  v8i r;
  r[0] = d0.x; r[1] = d0.y; r[2] = d1.x; r[3] = d1.y;
  r[4] = d2.x; r[5] = d2.y; r[6] = d3.x; r[7] = d3.y;
  return r;
}

__device__ __forceinline__ v8i load_b64x16(const signed char* base, int ldb, int lane) {
  const int n  = lane & 15;
  const int kh = (lane >> 4) << 3;
  const signed char* col = base + n * ldb + kh;
  int2 d0 = *(const int2*)(col);
  int2 d1 = *(const int2*)(col + 16);
  int2 d2 = *(const int2*)(col + 32);
  int2 d3 = *(const int2*)(col + 48);
  v8i r;
  r[0] = d0.x; r[1] = d0.y; r[2] = d1.x; r[3] = d1.y;
  r[4] = d2.x; r[5] = d2.y; r[6] = d3.x; r[7] = d3.y;
  return r;
}

__device__ __forceinline__ signed char sat8(float y) {
  int i = __float2int_rn(y);
  i = i > 127 ? 127 : (i < -128 ? -128 : i);
  return (signed char)i;
}

__global__ void pack_i8_kernel(const int* __restrict__ src, signed char* __restrict__ dst, int n) {
  int i = blockIdx.x * blockDim.x + threadIdx.x;
  if (i < n) dst[i] = (signed char)src[i];
}

__global__ void __launch_bounds__(256)
gemm_s8s8(const signed char* __restrict__ A, const signed char* __restrict__ B,
          const signed char* __restrict__ bias8, const float* __restrict__ biasf,
          signed char* __restrict__ out8, float* __restrict__ outf,
          int M, int N, int K, float alpha, int mode)
{
  const int lane  = threadIdx.x & 31;
  const int wave  = threadIdx.x >> 5;
  const int wm    = wave >> 2;
  const int wn    = wave & 3;
  const int bm    = blockIdx.y * 128 + wm * 64;
  const int bn    = blockIdx.x * 128 + wn * 32;
  const int nlane = lane & 15;
  const int mbase = (lane >> 4) << 3;

  v8i acc[4][2];
#pragma unroll
  for (int i = 0; i < 4; i++)
#pragma unroll
    for (int j = 0; j < 2; j++) acc[i][j] = zero8();

  for (int kb = 0; kb < K; kb += 64) {
    v8i a[4], b[2];
#pragma unroll
    for (int i = 0; i < 4; i++)
      a[i] = load_a16x64(A + (size_t)(bm + i * 16) * K + kb, K, lane);
#pragma unroll
    for (int j = 0; j < 2; j++)
      b[j] = load_b64x16(B + (size_t)(bn + j * 16) * K + kb, K, lane);

    if (kb + 64 < K) {
      __builtin_prefetch(A + (size_t)(bm + (lane & 3) * 16) * K + kb + 64, 0, 1);
      __builtin_prefetch(B + (size_t)(bn + (lane & 1) * 16) * K + kb + 64, 0, 1);
    }
#pragma unroll
    for (int i = 0; i < 4; i++)
#pragma unroll
      for (int j = 0; j < 2; j++)
        acc[i][j] = wmma_iu8(a[i], b[j], acc[i][j]);
  }

#pragma unroll
  for (int i = 0; i < 4; i++)
#pragma unroll
    for (int j = 0; j < 2; j++) {
      const int n = bn + j * 16 + nlane;
#pragma unroll
      for (int v = 0; v < 8; v++) {
        const int m = bm + i * 16 + mbase + v;
        const float y = alpha * (float)acc[i][j][v];
        if (mode == 2) {
          outf[(size_t)m * N + n] = y + biasf[n];
        } else {
          const signed char q = sat8(y + (float)bias8[n]);
          if (mode == 0) {
            out8[(size_t)m * N + n] = q;
          } else {
            const int bb = m >> 10, s = m & 1023, h = n >> 7, d = n & 127;
            out8[((size_t)((bb * H_NUM + h) * DH_DIM + d) << 10) + s] = q;
          }
        }
      }
    }
}

__global__ void __launch_bounds__(128)
attn_fused(const signed char* __restrict__ q8, const signed char* __restrict__ k8,
           const signed char* __restrict__ vt8, const float* __restrict__ mask,
           signed char* __restrict__ ctx8)
{
  extern __shared__ char smem[];
  float*       scores = (float*)smem;
  signed char* probs  = (signed char*)(smem + 32 * S_LEN * sizeof(float));

  const int lane  = threadIdx.x & 31;
  const int wave  = threadIdx.x >> 5;
  const int qt    = blockIdx.x;
  const int bh    = blockIdx.y;
  const int b     = bh >> 5, h = bh & 31;
  const int q0    = qt * 32;
  const int nlane = lane & 15;
  const int mbase = (lane >> 4) << 3;

  v8i aq[2][2];
#pragma unroll
  for (int mt = 0; mt < 2; mt++)
#pragma unroll
    for (int ks = 0; ks < 2; ks++)
      aq[mt][ks] = load_a16x64(
          q8 + (size_t)(b * S_LEN + q0 + mt * 16) * E_DIM + h * DH_DIM + ks * 64,
          E_DIM, lane);

  for (int jt = 0; jt < 16; jt++) {
    const int n16 = wave * 256 + jt * 16;
    const size_t krow = (size_t)(b * S_LEN + n16) * E_DIM + h * DH_DIM;
    v8i bk0 = load_b64x16(k8 + krow,      E_DIM, lane);
    v8i bk1 = load_b64x16(k8 + krow + 64, E_DIM, lane);
#pragma unroll
    for (int mt = 0; mt < 2; mt++) {
      v8i acc = zero8();
      acc = wmma_iu8(aq[mt][0], bk0, acc);
      acc = wmma_iu8(aq[mt][1], bk1, acc);
#pragma unroll
      for (int v = 0; v < 8; v++) {
        const int m  = mt * 16 + mbase + v;
        const int kk = n16 + nlane;
        float f = 3e-4f * (float)acc[v]
                + mask[(size_t)b * S_LEN * S_LEN + (size_t)(q0 + m) * S_LEN + kk];
        f = fmaxf(f, F32_MIN_CONST);
        scores[m * S_LEN + kk] = f;
      }
    }
  }
  __syncthreads();

  for (int r = wave * 8; r < wave * 8 + 8; r++) {
    float* row = scores + r * S_LEN;
    float mx = -3.4e38f;
    for (int c = lane; c < S_LEN; c += 32) mx = fmaxf(mx, row[c]);
#pragma unroll
    for (int off = 16; off > 0; off >>= 1) mx = fmaxf(mx, __shfl_xor(mx, off, 32));
    float sum = 0.f;
    for (int c = lane; c < S_LEN; c += 32) {
      float e = __expf(row[c] - mx);
      row[c] = e;
      sum += e;
    }
#pragma unroll
    for (int off = 16; off > 0; off >>= 1) sum += __shfl_xor(sum, off, 32);
    const float scale = 127.0f / sum;
    for (int c = lane; c < S_LEN; c += 32)
      probs[r * S_LEN + c] = sat8(row[c] * scale);
  }
  __syncthreads();

  v8i acc[2][2];
#pragma unroll
  for (int mt = 0; mt < 2; mt++)
#pragma unroll
    for (int nt = 0; nt < 2; nt++) acc[mt][nt] = zero8();

  for (int kb = 0; kb < S_LEN; kb += 64) {
    v8i ap[2], bv[2];
#pragma unroll
    for (int mt = 0; mt < 2; mt++)
      ap[mt] = load_a16x64((const signed char*)probs + mt * 16 * S_LEN + kb, S_LEN, lane);
#pragma unroll
    for (int nt = 0; nt < 2; nt++)
      bv[nt] = load_b64x16(
          vt8 + (size_t)(bh * DH_DIM + wave * 32 + nt * 16) * S_LEN + kb, S_LEN, lane);
#pragma unroll
    for (int mt = 0; mt < 2; mt++)
#pragma unroll
      for (int nt = 0; nt < 2; nt++)
        acc[mt][nt] = wmma_iu8(ap[mt], bv[nt], acc[mt][nt]);
  }

  const float pv_alpha = 1.0f / 127.0f;
#pragma unroll
  for (int mt = 0; mt < 2; mt++)
#pragma unroll
    for (int nt = 0; nt < 2; nt++) {
      const int d = wave * 32 + nt * 16 + nlane;
#pragma unroll
      for (int v = 0; v < 8; v++) {
        const int m = q0 + mt * 16 + mbase + v;
        ctx8[(size_t)(b * S_LEN + m) * E_DIM + h * DH_DIM + d] =
            sat8(pv_alpha * (float)acc[mt][nt][v]);
      }
    }
}

extern "C" void kernel_launch(void* const* d_in, const int* in_sizes, int n_in,
                              void* d_out, int out_size, void* d_ws, size_t ws_size,
                              hipStream_t stream) {
  const int*   hs_i32 = (const int*)d_in[0];
  const float* mask   = (const float*)d_in[1];
  const int*   wq_i32 = (const int*)d_in[2];
  const int*   bq_i32 = (const int*)d_in[3];
  const int*   wk_i32 = (const int*)d_in[4];
  const int*   bk_i32 = (const int*)d_in[5];
  const int*   wv_i32 = (const int*)d_in[6];
  const int*   bv_i32 = (const int*)d_in[7];
  const int*   wo_i32 = (const int*)d_in[8];
  const float* b_out  = (const float*)d_in[9];
  float* out = (float*)d_out;

  char* ws = (char*)d_ws;
  size_t off = 0;
  auto alloc = [&](size_t bytes) {
    char* p = ws + off;
    off += (bytes + 255) & ~(size_t)255;
    return p;
  };
  signed char* hs  = (signed char*)alloc((size_t)M_ROWS * E_DIM);
  signed char* wq  = (signed char*)alloc((size_t)E_DIM * E_DIM);
  signed char* wk  = (signed char*)alloc((size_t)E_DIM * E_DIM);
  signed char* wv  = (signed char*)alloc((size_t)E_DIM * E_DIM);
  signed char* wo  = (signed char*)alloc((size_t)E_DIM * E_DIM);
  signed char* bq  = (signed char*)alloc(E_DIM);
  signed char* bk  = (signed char*)alloc(E_DIM);
  signed char* bv  = (signed char*)alloc(E_DIM);
  signed char* q8  = (signed char*)alloc((size_t)M_ROWS * E_DIM);
  signed char* k8  = (signed char*)alloc((size_t)M_ROWS * E_DIM);
  signed char* vt  = (signed char*)alloc((size_t)M_ROWS * E_DIM);
  signed char* ctx = (signed char*)alloc((size_t)M_ROWS * E_DIM);

  auto pack = [&](const int* s, signed char* d, int n) {
    pack_i8_kernel<<<(n + 255) / 256, 256, 0, stream>>>(s, d, n);
  };
  pack(hs_i32, hs, M_ROWS * E_DIM);
  pack(wq_i32, wq, E_DIM * E_DIM);
  pack(wk_i32, wk, E_DIM * E_DIM);
  pack(wv_i32, wv, E_DIM * E_DIM);
  pack(wo_i32, wo, E_DIM * E_DIM);
  pack(bq_i32, bq, E_DIM);
  pack(bk_i32, bk, E_DIM);
  pack(bv_i32, bv, E_DIM);

  const dim3 ggrid(E_DIM / 128, M_ROWS / 128);
  gemm_s8s8<<<ggrid, 256, 0, stream>>>(hs, wq, bq, nullptr, q8, nullptr,
                                       M_ROWS, E_DIM, E_DIM, 2e-4f, 0);
  gemm_s8s8<<<ggrid, 256, 0, stream>>>(hs, wk, bk, nullptr, k8, nullptr,
                                       M_ROWS, E_DIM, E_DIM, 2e-4f, 0);
  gemm_s8s8<<<ggrid, 256, 0, stream>>>(hs, wv, bv, nullptr, vt, nullptr,
                                       M_ROWS, E_DIM, E_DIM, 2e-4f, 1);

  const size_t lds_bytes = 32 * S_LEN * sizeof(float) + 32 * S_LEN;
  attn_fused<<<dim3(S_LEN / 32, 2 * H_NUM), 128, lds_bytes, stream>>>(q8, k8, vt, mask, ctx);

  gemm_s8s8<<<ggrid, 256, 0, stream>>>(ctx, wo, nullptr, b_out, nullptr, out,
                                       M_ROWS, E_DIM, E_DIM, 2e-4f, 2);
}
